// ResidualAttentionBlock_44976897523741
// MI455X (gfx1250) — matrix operand order repaired and checked by reading, not compiled, not run on hardware
//
#include <hip/hip_runtime.h>
#include <hip/hip_bf16.h>
#include <stdint.h>

typedef __attribute__((ext_vector_type(16))) _Float16 v16h;
typedef __attribute__((ext_vector_type(8)))  float    v8f;
typedef __attribute__((ext_vector_type(4)))  int      v4i;

#define Nb 4
#define Qq 8
#define Dd 1024
#define Hh 16
#define Cc 64
#define KK 2056
#define PP 257
#define KP 2080          // keys padded: 65*32, 130*16
#define STATS_LD 640     // [0]=Mp [1]=Sp [2..9]=Mf [10..17]=Sf [32..288]=Mt [352..608]=St
#define LN_EPS 1e-5f
#define SCALE 0.125f     // 1/sqrt(64)

#if __has_builtin(__builtin_amdgcn_global_load_async_to_lds_b128)
#define HAVE_ASYNC_LDS 1
#else
#define HAVE_ASYNC_LDS 0
#endif

// ---------------- async global -> LDS (gfx1250) ----------------

__device__ __forceinline__ void async_b128(const float* g, float* l) {
#if HAVE_ASYNC_LDS
  __builtin_amdgcn_global_load_async_to_lds_b128((v4i*)g, (v4i*)l, 0, 0);
#else
  (void)g; (void)l;
#endif
}
__device__ __forceinline__ void async_wait() {
#if HAVE_ASYNC_LDS
  asm volatile("s_wait_asynccnt 0x0" ::: "memory");
#endif
}

// ---------------- WMMA helpers (contiguous k order, same in A and B) ----------------

__device__ __forceinline__ v8f wmma16(v16h a, v16h b, v8f c) {
  return __builtin_amdgcn_wmma_f32_16x16x32_f16(false, a, false, b, (short)0, c, false, false);
}

// A fragment: lane&15 = M row; element i -> K offset i + 16*(lane>>4)
__device__ __forceinline__ int a_koff(int i, int hf) { return i + hf * 16; }

__device__ __forceinline__ v16h frag_a_f16g(const _Float16* A, int lda, int m0, int k0) {
  int lane = threadIdx.x & 31, hf = lane >> 4, m = m0 + (lane & 15);
  v16h r;
#pragma unroll
  for (int i = 0; i < 16; ++i) r[i] = A[(size_t)m * lda + k0 + a_koff(i, hf)];
  return r;
}

// B fragment from transposed f16 weights WT[N][K]: lane&15 = N col,
// element i -> K offset i + 16*(lane>>4)  (contiguous 32B per lane -> b128 loads)
__device__ __forceinline__ v16h frag_bT16(const _Float16* BT, int ldk, int k0, int n0) {
  int lane = threadIdx.x & 31, hf = lane >> 4, nn = n0 + (lane & 15);
  v16h r;
#pragma unroll
  for (int i = 0; i < 16; ++i) r[i] = BT[(size_t)nn * ldk + k0 + i + hf * 16];
  return r;
}

// B[kc][key] fragment where LDS holds f32 tile S[key][kc] (key-major, ld=Cc)
__device__ __forceinline__ v16h frag_bT_ldsKf(const float* S, int c0) {
  int lane = threadIdx.x & 31, hf = lane >> 4, key = lane & 15;
  v16h r;
#pragma unroll
  for (int i = 0; i < 16; ++i) r[i] = (_Float16)S[key * Cc + c0 + i + hf * 16];
  return r;
}

// B[kk][c] fragment from f32 LDS tile S[kk][c] (row-major, ld=Cc)
__device__ __forceinline__ v16h frag_b_ldsVf(const float* S, int n0) {
  int lane = threadIdx.x & 31, hf = lane >> 4, nn = n0 + (lane & 15);
  v16h r;
#pragma unroll
  for (int i = 0; i < 16; ++i) r[i] = (_Float16)S[(i + hf * 16) * Cc + nn];
  return r;
}

// q fragment: [8][64] f16, rows 8..15 zero-padded
__device__ __forceinline__ v16h frag_q(const _Float16* Qm, int c0) {
  int lane = threadIdx.x & 31, hf = lane >> 4, m = lane & 15;
  v16h r;
#pragma unroll
  for (int i = 0; i < 16; ++i) {
    int c = c0 + a_koff(i, hf);
    r[i] = (m < Qq) ? Qm[m * Cc + c] : (_Float16)0.0f;
  }
  return r;
}

// ---------------- transpose + convert: W[K][N] f32 -> WT[N][K] f16 ----------------

__global__ __launch_bounds__(256) void wtrans_kernel(const float* __restrict__ W,
                                                     _Float16* __restrict__ WT,
                                                     int Kdim, int Ndim) {
  __shared__ _Float16 t[32][33];
  int n0 = blockIdx.x * 32, k0 = blockIdx.y * 32;
  int tx = threadIdx.x & 31, ty = threadIdx.x >> 5;
  for (int r = ty; r < 32; r += 8)
    t[r][tx] = (_Float16)W[(size_t)(k0 + r) * Ndim + n0 + tx];
  __syncthreads();
  for (int r = ty; r < 32; r += 8)
    WT[(size_t)(n0 + r) * Kdim + k0 + tx] = t[tx][r];
}

// ---------------- LayerNorm row -> f16 ----------------

__global__ __launch_bounds__(256) void ln_kernel(const float* __restrict__ x,
                                                 const float* __restrict__ w,
                                                 const float* __restrict__ b,
                                                 _Float16* __restrict__ y) {
  int row = blockIdx.x;
  const float* xr = x + (size_t)row * Dd;
  int tid = threadIdx.x, lane = tid & 31, wv = tid >> 5;
  float s = 0.f, s2 = 0.f;
  for (int i = tid; i < Dd; i += 256) { float v = xr[i]; s += v; s2 += v * v; }
#pragma unroll
  for (int o = 16; o >= 1; o >>= 1) { s += __shfl_xor(s, o, 32); s2 += __shfl_xor(s2, o, 32); }
  __shared__ float rs[8], rs2[8], mu_s, inv_s;
  if (lane == 0) { rs[wv] = s; rs2[wv] = s2; }
  __syncthreads();
  if (tid == 0) {
    float S = 0.f, S2 = 0.f;
    for (int k = 0; k < 8; ++k) { S += rs[k]; S2 += rs2[k]; }
    float mu = S / (float)Dd;
    float var = S2 / (float)Dd - mu * mu;
    mu_s = mu; inv_s = rsqrtf(var + LN_EPS);
  }
  __syncthreads();
  float mu = mu_s, inv = inv_s;
  for (int i = tid; i < Dd; i += 256)
    y[(size_t)row * Dd + i] = (_Float16)((xr[i] - mu) * inv * w[i] + b[i]);
}

// ---------------- in-projection GEMM (32x2048x1024, WMMA) ----------------

__global__ __launch_bounds__(32) void inproj_kernel(const _Float16* __restrict__ xln,
                                                    const _Float16* __restrict__ WiT,
                                                    const float* __restrict__ bi,
                                                    _Float16* __restrict__ qs16,
                                                    _Float16* __restrict__ qc16,
                                                    float* __restrict__ qc32) {
  int nt = blockIdx.x, mt = blockIdx.y;
  v8f acc = {};
  for (int ks = 0; ks < Dd / 32; ++ks) {
    v16h a = frag_a_f16g(xln, Dd, mt * 16, ks * 32);
    v16h bb = frag_bT16(WiT, Dd, ks * 32, nt * 16);
    acc = wmma16(a, bb, acc);
  }
  int lane = threadIdx.x & 31, hf = lane >> 4, nn = lane & 15;
  int col = nt * 16 + nn;
  int h = col >> 7, rem = col & 127, act = rem >> 6, c = rem & 63;
#pragma unroll
  for (int r = 0; r < 8; ++r) {
    int row = mt * 16 + r + hf * 8;
    int n = row >> 3, q = row & 7;
    float v = acc[r] + bi[col];
    size_t qo = (((size_t)(n * Hh + h)) * Qq + q) * Cc + c;
    if (act == 0) {
      qs16[qo] = (_Float16)(v * SCALE);
    } else {
      qc16[qo] = (_Float16)(v * SCALE);
      qc32[qo] = v;
    }
  }
}

// ---------------- affinity + coda-tanh via WMMA, per (n,h) ----------------

__global__ __launch_bounds__(256) void attn_aff_kernel(const _Float16* __restrict__ qs16,
                                                       const _Float16* __restrict__ qc16,
                                                       const float* __restrict__ Kv,
                                                       const uint8_t* __restrict__ msk,
                                                       float* __restrict__ aff,
                                                       float* __restrict__ cab) {
  int nh = blockIdx.x, n = nh / Hh, h = nh % Hh;
  int tid = threadIdx.x, lane = tid & 31, wv = tid >> 5;
  __shared__ float kst[8][16 * Cc];   // per-wave 16-key x 64-ch f32 tile (32KB)
  const _Float16* Qs = qs16 + (size_t)nh * Qq * Cc;
  const _Float16* Qc = qc16 + (size_t)nh * Qq * Cc;
  v16h a0 = frag_q(Qs, 0), a1 = frag_q(Qs, 32);
  v16h q0 = frag_q(Qc, 0), q1 = frag_q(Qc, 32);
  float* arow = aff + (size_t)nh * 16 * KP;
  float* crow = cab + (size_t)nh * 16 * KP;
  for (int t = wv; t < KP / 16; t += 8) {
    int kbase = t * 16;
    float* ks = kst[wv];
#if HAVE_ASYNC_LDS
#pragma unroll
    for (int it = 0; it < 8; ++it) {
      int idx4 = it * 32 + lane, key = idx4 >> 4, c4 = (idx4 & 15) << 2;
      int gk = kbase + key;
      if (gk < KK)
        async_b128(Kv + ((((size_t)n * KK + gk) * Hh + h) * Cc + c4), ks + key * Cc + c4);
    }
    async_wait();
    if (kbase + 15 >= KK) {
#pragma unroll
      for (int it = 0; it < 8; ++it) {
        int idx4 = it * 32 + lane, key = idx4 >> 4, c4 = (idx4 & 15) << 2;
        if (kbase + key >= KK) {
          ks[key * Cc + c4 + 0] = 0.f; ks[key * Cc + c4 + 1] = 0.f;
          ks[key * Cc + c4 + 2] = 0.f; ks[key * Cc + c4 + 3] = 0.f;
        }
      }
    }
#else
#pragma unroll
    for (int it = 0; it < 8; ++it) {
      int idx4 = it * 32 + lane, key = idx4 >> 4, c4 = (idx4 & 15) << 2;
      int gk = kbase + key;
      float4 v = {0.f, 0.f, 0.f, 0.f};
      if (gk < KK) v = *(const float4*)(Kv + ((((size_t)n * KK + gk) * Hh + h) * Cc + c4));
      ks[key * Cc + c4 + 0] = v.x; ks[key * Cc + c4 + 1] = v.y;
      ks[key * Cc + c4 + 2] = v.z; ks[key * Cc + c4 + 3] = v.w;
    }
#endif
    v16h b0 = frag_bT_ldsKf(ks, 0), b1 = frag_bT_ldsKf(ks, 32);
    v8f ac = {}; ac = wmma16(a0, b0, ac); ac = wmma16(a1, b1, ac);
    v8f cc = {}; cc = wmma16(q0, b0, cc); cc = wmma16(q1, b1, cc);
    int hf = lane >> 4, nn = lane & 15;
    int gkn = kbase + nn;
    bool mv = (gkn < KK) && (msk[(size_t)n * KK + gkn] != 0);
#pragma unroll
    for (int r = 0; r < 8; ++r) {
      int mrow = r + hf * 8;
      arow[(size_t)mrow * KP + gkn] = mv ? ac[r] : -1e30f;
      crow[(size_t)mrow * KP + gkn] = tanhf(cc[r]);
    }
  }
}

// ---------------- softmax statistics (patch/frame/temporal) ----------------

__global__ __launch_bounds__(256) void stats_kernel(const float* __restrict__ aff,
                                                    float* __restrict__ st) {
  int b = blockIdx.x;                 // b = nh*Qq + q
  int q = b % Qq, nh = b / Qq;
  const float* row = aff + ((size_t)nh * 16 + q) * KP;
  float* s = st + (size_t)b * STATS_LD;
  int tid = threadIdx.x, lane = tid & 31, wv = tid >> 5;
  __shared__ float Mf[8], Sf[8];
  {
    const float* fr = row + wv * PP;
    float mx = -1e30f;
    for (int j = lane; j < PP; j += 32) mx = fmaxf(mx, fr[j]);
#pragma unroll
    for (int o = 16; o >= 1; o >>= 1) mx = fmaxf(mx, __shfl_xor(mx, o, 32));
    float su = 0.f;
    for (int j = lane; j < PP; j += 32) su += __expf(fr[j] - mx);
#pragma unroll
    for (int o = 16; o >= 1; o >>= 1) su += __shfl_xor(su, o, 32);
    if (lane == 0) { Mf[wv] = mx; Sf[wv] = su; s[2 + wv] = mx; s[10 + wv] = su; }
  }
  __syncthreads();
  if (tid == 0) {
    float mp = -1e30f;
    for (int f = 0; f < 8; ++f) mp = fmaxf(mp, Mf[f]);
    float sp = 0.f;
    for (int f = 0; f < 8; ++f) sp += Sf[f] * __expf(Mf[f] - mp);
    s[0] = mp; s[1] = sp;
  }
  for (int p = tid; p < PP; p += 256) {
    float mx = -1e30f;
#pragma unroll
    for (int f = 0; f < 8; ++f) mx = fmaxf(mx, row[f * PP + p]);
    float su = 0.f;
#pragma unroll
    for (int f = 0; f < 8; ++f) su += __expf(row[f * PP + p] - mx);
    s[32 + p] = mx; s[352 + p] = su;
  }
}

// ---------------- L1 gate + triple-softmax combine (writes f16 aff_total) ----------------

__global__ __launch_bounds__(256) void combine_kernel(const float* __restrict__ qc32,
                                                      const float* __restrict__ Kv,
                                                      const uint8_t* __restrict__ msk,
                                                      const float* __restrict__ aff,
                                                      const float* __restrict__ cab,
                                                      const float* __restrict__ st,
                                                      _Float16* __restrict__ tot) {
  int nh = blockIdx.y, n = nh / Hh, h = nh % Hh;
  __shared__ float qcs[Qq * Cc];
  for (int i = threadIdx.x; i < Qq * Cc; i += 256) qcs[i] = qc32[(size_t)nh * Qq * Cc + i];
  __syncthreads();
  int k = blockIdx.x * 256 + threadIdx.x;
  if (k >= KP) return;
  _Float16* trow = tot + (size_t)nh * 16 * KP;
  if (k >= KK) {
    for (int mr = 0; mr < 16; ++mr) trow[(size_t)mr * KP + k] = (_Float16)0.f;
    return;
  }
  bool mv = msk[(size_t)n * KK + k] != 0;
  float kr[Cc];
  const float* kp = Kv + (((size_t)n * KK + k) * Hh + h) * Cc;
#pragma unroll
  for (int c4 = 0; c4 < 16; ++c4) {
    float4 v = *(const float4*)(kp + c4 * 4);
    kr[c4 * 4 + 0] = v.x; kr[c4 * 4 + 1] = v.y;
    kr[c4 * 4 + 2] = v.z; kr[c4 * 4 + 3] = v.w;
  }
  int f = k / PP, p = k - f * PP;
  const float* arow = aff + (size_t)nh * 16 * KP;
  const float* crow = cab + (size_t)nh * 16 * KP;
  for (int q = 0; q < Qq; ++q) {
    const float* sq = st + ((size_t)nh * Qq + q) * STATS_LD;
    float a = arow[(size_t)q * KP + k];
    float ca = crow[(size_t)q * KP + k];
    float l1 = 0.f;
#pragma unroll
    for (int c = 0; c < Cc; ++c) l1 += fabsf(qcs[q * Cc + c] - kr[c]);
    float g = mv ? 2.f / (1.f + __expf(l1 * SCALE)) : 0.f;
    float smx = (__expf(a - sq[0]) / sq[1] +
                 __expf(a - sq[2 + f]) / sq[10 + f] +
                 __expf(a - sq[32 + p]) / sq[352 + p]) * (1.f / 3.f);
    trow[(size_t)q * KP + k] = (_Float16)(0.5f * (smx + ca * g));
  }
  for (int mr = Qq; mr < 16; ++mr) trow[(size_t)mr * KP + k] = (_Float16)0.f;
}

// ---------------- mix = aff_total @ V via WMMA, per (n,h) ----------------

__global__ __launch_bounds__(256) void mix_kernel(const _Float16* __restrict__ tot,
                                                  const float* __restrict__ Vv,
                                                  _Float16* __restrict__ mixb) {
  int nh = blockIdx.x, n = nh / Hh, h = nh % Hh;
  int tid = threadIdx.x, lane = tid & 31, wv = tid >> 5;
  __shared__ float vst[8][32 * Cc];    // 64KB: per-wave 32-key x 64-ch f32 V tile
  __shared__ float wsum[8][16 * Cc];   // 32KB: per-wave partial outputs
  const _Float16* trow = tot + (size_t)nh * 16 * KP;
  v8f acc[4] = {};
  for (int t = wv; t < KP / 32; t += 8) {
    int kbase = t * 32;
    float* vs = vst[wv];
#if HAVE_ASYNC_LDS
#pragma unroll
    for (int it = 0; it < 16; ++it) {
      int idx4 = it * 32 + lane, key = idx4 >> 4, c4 = (idx4 & 15) << 2;
      int gk = kbase + key;
      if (gk < KK)
        async_b128(Vv + ((((size_t)n * KK + gk) * Hh + h) * Cc + c4), vs + key * Cc + c4);
    }
    async_wait();
    if (kbase + 31 >= KK) {
#pragma unroll
      for (int it = 0; it < 16; ++it) {
        int idx4 = it * 32 + lane, key = idx4 >> 4, c4 = (idx4 & 15) << 2;
        if (kbase + key >= KK) {
          vs[key * Cc + c4 + 0] = 0.f; vs[key * Cc + c4 + 1] = 0.f;
          vs[key * Cc + c4 + 2] = 0.f; vs[key * Cc + c4 + 3] = 0.f;
        }
      }
    }
#else
#pragma unroll
    for (int it = 0; it < 16; ++it) {
      int idx4 = it * 32 + lane, key = idx4 >> 4, c4 = (idx4 & 15) << 2;
      int gk = kbase + key;
      float4 v = {0.f, 0.f, 0.f, 0.f};
      if (gk < KK) v = *(const float4*)(Vv + ((((size_t)n * KK + gk) * Hh + h) * Cc + c4));
      vs[key * Cc + c4 + 0] = v.x; vs[key * Cc + c4 + 1] = v.y;
      vs[key * Cc + c4 + 2] = v.z; vs[key * Cc + c4 + 3] = v.w;
    }
#endif
    v16h af = frag_a_f16g(trow, KP, 0, kbase);
#pragma unroll
    for (int ct = 0; ct < 4; ++ct) {
      v16h bf = frag_b_ldsVf(vs, ct * 16);
      acc[ct] = wmma16(af, bf, acc[ct]);
    }
  }
  int hf = lane >> 4, nn = lane & 15;
#pragma unroll
  for (int ct = 0; ct < 4; ++ct)
#pragma unroll
    for (int r = 0; r < 8; ++r)
      wsum[wv][(r + hf * 8) * Cc + ct * 16 + nn] = acc[ct][r];
  __syncthreads();
  for (int i = tid; i < Qq * Cc; i += 256) {
    int q = i >> 6, c = i & 63;
    float sm = 0.f;
#pragma unroll
    for (int w8 = 0; w8 < 8; ++w8) sm += wsum[w8][q * Cc + c];
    mixb[((size_t)(n * Qq + q)) * Dd + h * Cc + c] = (_Float16)sm;
  }
}

// ---------------- out-projection + residual (WMMA) ----------------

__global__ __launch_bounds__(32) void outproj_kernel(const _Float16* __restrict__ mixb,
                                                     const _Float16* __restrict__ WT,
                                                     const float* __restrict__ bias,
                                                     const float* __restrict__ xres,
                                                     float* __restrict__ y) {
  int nt = blockIdx.x, mt = blockIdx.y;
  v8f acc = {};
  for (int ks = 0; ks < Dd / 32; ++ks) {
    v16h a = frag_a_f16g(mixb, Dd, mt * 16, ks * 32);
    v16h bb = frag_bT16(WT, Dd, ks * 32, nt * 16);
    acc = wmma16(a, bb, acc);
  }
  int lane = threadIdx.x & 31, hf = lane >> 4, nn = lane & 15;
  int col = nt * 16 + nn;
#pragma unroll
  for (int r = 0; r < 8; ++r) {
    int row = mt * 16 + r + hf * 8;
    y[(size_t)row * Dd + col] = acc[r] + bias[col] + xres[(size_t)row * Dd + col];
  }
}

// ---------------- fc + quick-GELU (WMMA) ----------------

__global__ __launch_bounds__(32) void fc_kernel(const _Float16* __restrict__ hln,
                                                const _Float16* __restrict__ WT,
                                                const float* __restrict__ bias,
                                                _Float16* __restrict__ g16) {
  int nt = blockIdx.x, mt = blockIdx.y;
  int lane = threadIdx.x & 31;
  v8f acc = {};
  for (int ks = 0; ks < Dd / 32; ++ks) {
    __builtin_prefetch(WT + (size_t)(nt * 16 + (lane & 15)) * Dd + ks * 32 + 128, 0, 0);
    v16h a = frag_a_f16g(hln, Dd, mt * 16, ks * 32);
    v16h bb = frag_bT16(WT, Dd, ks * 32, nt * 16);
    acc = wmma16(a, bb, acc);
  }
  int hf = lane >> 4, nn = lane & 15;
  int col = nt * 16 + nn;
#pragma unroll
  for (int r = 0; r < 8; ++r) {
    int row = mt * 16 + r + hf * 8;
    float v = acc[r] + bias[col];
    float gel = v / (1.f + __expf(-1.702f * v));
    g16[(size_t)row * (4 * Dd) + col] = (_Float16)gel;
  }
}

// ---------------- MLP projection + residual (WMMA) ----------------

__global__ __launch_bounds__(32) void projout_kernel(const _Float16* __restrict__ g16,
                                                     const _Float16* __restrict__ WT,
                                                     const float* __restrict__ bias,
                                                     const float* __restrict__ xres,
                                                     float* __restrict__ y) {
  int nt = blockIdx.x, mt = blockIdx.y;
  int lane = threadIdx.x & 31;
  v8f acc = {};
  for (int ks = 0; ks < (4 * Dd) / 32; ++ks) {
    __builtin_prefetch(WT + (size_t)(nt * 16 + (lane & 15)) * (4 * Dd) + ks * 32 + 128, 0, 0);
    v16h a = frag_a_f16g(g16, 4 * Dd, mt * 16, ks * 32);
    v16h bb = frag_bT16(WT, 4 * Dd, ks * 32, nt * 16);
    acc = wmma16(a, bb, acc);
  }
  int hf = lane >> 4, nn = lane & 15;
  int col = nt * 16 + nn;
#pragma unroll
  for (int r = 0; r < 8; ++r) {
    int row = mt * 16 + r + hf * 8;
    y[(size_t)row * Dd + col] = acc[r] + bias[col] + xres[(size_t)row * Dd + col];
  }
}

// ---------------- Host launcher ----------------

extern "C" void kernel_launch(void* const* d_in, const int* in_sizes, int n_in,
                              void* d_out, int out_size, void* d_ws, size_t ws_size,
                              hipStream_t stream) {
  (void)in_sizes; (void)n_in; (void)out_size; (void)ws_size;
  const float*   x    = (const float*)d_in[0];
  const float*   Kv   = (const float*)d_in[1];
  const float*   Vv   = (const float*)d_in[2];
  const uint8_t* msk  = (const uint8_t*)d_in[3];
  const float*   ln1w = (const float*)d_in[4];
  const float*   ln1b = (const float*)d_in[5];
  const float*   inw  = (const float*)d_in[6];
  const float*   inb  = (const float*)d_in[7];
  const float*   ow   = (const float*)d_in[8];
  const float*   ob   = (const float*)d_in[9];
  const float*   ln2w = (const float*)d_in[10];
  const float*   ln2b = (const float*)d_in[11];
  const float*   fcw  = (const float*)d_in[12];
  const float*   fcb  = (const float*)d_in[13];
  const float*   pw   = (const float*)d_in[14];
  const float*   pb   = (const float*)d_in[15];
  float* out = (float*)d_out;

  char* ws = (char*)d_ws;
  size_t off = 0;
  auto take = [&](size_t bytes) -> void* {
    off = (off + 255) & ~(size_t)255;
    void* p = ws + off;
    off += bytes;
    return p;
  };
  _Float16* xln  = (_Float16*)take((size_t)32 * Dd * 2);
  _Float16* qs16 = (_Float16*)take((size_t)Nb * Hh * Qq * Cc * 2);
  _Float16* qc16 = (_Float16*)take((size_t)Nb * Hh * Qq * Cc * 2);
  float*    qc32 = (float*)   take((size_t)Nb * Hh * Qq * Cc * 4);
  float*    aff  = (float*)   take((size_t)Nb * Hh * 16 * KP * 4);
  float*    cab  = (float*)   take((size_t)Nb * Hh * 16 * KP * 4);
  _Float16* tot  = (_Float16*)take((size_t)Nb * Hh * 16 * KP * 2);
  float*    st   = (float*)   take((size_t)Nb * Hh * Qq * STATS_LD * 4);
  _Float16* mixb = (_Float16*)take((size_t)32 * Dd * 2);
  float*    x1   = (float*)   take((size_t)32 * Dd * 4);
  _Float16* h16  = (_Float16*)take((size_t)32 * Dd * 2);
  _Float16* g16  = (_Float16*)take((size_t)32 * 4 * Dd * 2);
  _Float16* inT  = (_Float16*)take((size_t)2 * Dd * Dd * 2);      // [2048][1024]
  _Float16* owT  = (_Float16*)take((size_t)Dd * Dd * 2);          // [1024][1024]
  _Float16* fcT  = (_Float16*)take((size_t)4 * Dd * Dd * 2);      // [4096][1024]
  _Float16* pwT  = (_Float16*)take((size_t)Dd * 4 * Dd * 2);      // [1024][4096]

  // weight prep (transpose + f16 convert), L2-resident
  wtrans_kernel<<<dim3(2 * Dd / 32, Dd / 32), 256, 0, stream>>>(inw, inT, Dd, 2 * Dd);
  wtrans_kernel<<<dim3(Dd / 32, Dd / 32), 256, 0, stream>>>(ow, owT, Dd, Dd);
  wtrans_kernel<<<dim3(4 * Dd / 32, Dd / 32), 256, 0, stream>>>(fcw, fcT, Dd, 4 * Dd);
  wtrans_kernel<<<dim3(Dd / 32, 4 * Dd / 32), 256, 0, stream>>>(pw, pwT, 4 * Dd, Dd);

  ln_kernel<<<32, 256, 0, stream>>>(x, ln1w, ln1b, xln);
  inproj_kernel<<<dim3(128, 2), 32, 0, stream>>>(xln, inT, inb, qs16, qc16, qc32);
  attn_aff_kernel<<<Nb * Hh, 256, 0, stream>>>(qs16, qc16, Kv, msk, aff, cab);
  stats_kernel<<<Nb * Hh * Qq, 256, 0, stream>>>(aff, st);
  combine_kernel<<<dim3(9, Nb * Hh), 256, 0, stream>>>(qc32, Kv, msk, aff, cab, st, tot);
  mix_kernel<<<Nb * Hh, 256, 0, stream>>>(tot, Vv, mixb);
  outproj_kernel<<<dim3(64, 2), 32, 0, stream>>>(mixb, owT, ob, x, x1);
  ln_kernel<<<32, 256, 0, stream>>>(x1, ln2w, ln2b, h16);
  fc_kernel<<<dim3(256, 2), 32, 0, stream>>>(h16, fcT, fcb, g16);
  projout_kernel<<<dim3(64, 2), 32, 0, stream>>>(g16, pwT, pb, x1, out);
}
